// Seq2Seq_57501022159056
// MI455X (gfx1250) — matrix operand order repaired and checked by reading, not compiled, not run on hardware
//
#include <hip/hip_runtime.h>
#include <hip/hip_bf16.h>

typedef _Float16 half_t;
typedef __attribute__((ext_vector_type(16))) _Float16 v16h;
typedef __attribute__((ext_vector_type(8)))  _Float16 v8h;
typedef __attribute__((ext_vector_type(8)))  float    v8f;

#define Bsz 64
#define Ssz 128
#define Tsz 128
#define Esz 128
#define Hsz 256
#define Vout 32000

// ---------------- WMMA fragment loaders (f16, 16x16x32) ----------------
// A (16x32, MxK): lane L -> row M = L%16; halfs j=0..15 map to
//   K = k0 + (L/16)*16 + j  => 16 contiguous halfs (two 8-half chunks).
__device__ __forceinline__ v16h load_frag_a(const half_t* p, int ld, int m0, int k0, int lane) {
  union { v16h v; v8h h[2]; } u;
  const half_t* base = p + (size_t)(m0 + (lane & 15)) * ld + k0 + ((lane >> 4) << 4);
  u.h[0] = *(const v8h*)(base);
  u.h[1] = *(const v8h*)(base + 8);
  return u.v;
}

// B (32x16, KxN) from row-major W[N][K] (B = W^T): lane L -> col N = L%16;
//   halfs j=0..15 map to K = k0 + (L/16)*16 + j  => one 32B contiguous load.
__device__ __forceinline__ v16h load_frag_b(const half_t* p, int ld, int n0, int k0, int lane) {
  const half_t* base = p + (size_t)(n0 + (lane & 15)) * ld + k0 + ((lane >> 4) << 4);
  return *(const v16h*)(base);
}

__device__ __forceinline__ float fast_rcp(float x) { return __builtin_amdgcn_rcpf(x); }
__device__ __forceinline__ float sigmoid_f(float x) {
  return fast_rcp(1.f + __expf(-x));
}
__device__ __forceinline__ float tanh_f(float x) {
  x = fminf(fmaxf(x, -15.f), 15.f);
  float e = __expf(-2.f * x);
  return (1.f - e) * fast_rcp(1.f + e);
}

// ---------------- persistent recurrence: one LSTM layer step ----------------
// 64 tasks = 4 Mtiles x 16 Ntiles(of H=256); wave w owns tasks 2w, 2w+1.
// Each task computes i/f/g/o 16x16 gate tiles for the same (b,n) block, so the
// cell update is fully in-lane; c-state is carried in registers across steps.
// Bias sums (bih+bhh, time-invariant) come from a 4KB LDS table.
template<int KX>
__device__ __forceinline__ void layer_step(
    const half_t* __restrict__ xin,   // [64][KX]   (LDS)
    const half_t* __restrict__ hin,   // [64][256]  (LDS, prev h)
    const half_t* __restrict__ Wih,   // [1024][KX] (global f16)
    const half_t* __restrict__ Whh,   // [1024][256](global f16)
    const float*  __restrict__ sbias, // [1024] f32 (LDS, bih+bhh)
    float cst[2][8],
    half_t* __restrict__ hout,        // [64][256]  (LDS, new h)
    half_t* __restrict__ gdump,       // optional global [64][256] f16
    int wave, int lane)
{
#pragma unroll
  for (int task = 0; task < 2; ++task) {
    int tt = wave * 2 + task;
    int mt = tt & 3;                  // batch tile (16 rows)
    int nh = tt >> 2;                 // H column tile (16 cols)
    v8f a0 = {}, a1 = {}, a2 = {}, a3 = {};
    for (int k0 = 0; k0 < KX; k0 += 32) {
      v16h a  = load_frag_a(xin, KX, mt * 16, k0, lane);
      v16h b0 = load_frag_b(Wih, KX, 0 * 256 + nh * 16, k0, lane);
      v16h b1 = load_frag_b(Wih, KX, 1 * 256 + nh * 16, k0, lane);
      v16h b2 = load_frag_b(Wih, KX, 2 * 256 + nh * 16, k0, lane);
      v16h b3 = load_frag_b(Wih, KX, 3 * 256 + nh * 16, k0, lane);
      a0 = __builtin_amdgcn_wmma_f32_16x16x32_f16(false, a, false, b0, (short)0, a0, false, false);
      a1 = __builtin_amdgcn_wmma_f32_16x16x32_f16(false, a, false, b1, (short)0, a1, false, false);
      a2 = __builtin_amdgcn_wmma_f32_16x16x32_f16(false, a, false, b2, (short)0, a2, false, false);
      a3 = __builtin_amdgcn_wmma_f32_16x16x32_f16(false, a, false, b3, (short)0, a3, false, false);
    }
    for (int k0 = 0; k0 < Hsz; k0 += 32) {
      v16h a  = load_frag_a(hin, Hsz, mt * 16, k0, lane);
      v16h b0 = load_frag_b(Whh, Hsz, 0 * 256 + nh * 16, k0, lane);
      v16h b1 = load_frag_b(Whh, Hsz, 1 * 256 + nh * 16, k0, lane);
      v16h b2 = load_frag_b(Whh, Hsz, 2 * 256 + nh * 16, k0, lane);
      v16h b3 = load_frag_b(Whh, Hsz, 3 * 256 + nh * 16, k0, lane);
      a0 = __builtin_amdgcn_wmma_f32_16x16x32_f16(false, a, false, b0, (short)0, a0, false, false);
      a1 = __builtin_amdgcn_wmma_f32_16x16x32_f16(false, a, false, b1, (short)0, a1, false, false);
      a2 = __builtin_amdgcn_wmma_f32_16x16x32_f16(false, a, false, b2, (short)0, a2, false, false);
      a3 = __builtin_amdgcn_wmma_f32_16x16x32_f16(false, a, false, b3, (short)0, a3, false, false);
    }
    int ncol = nh * 16 + (lane & 15);                 // column within H
    float bi = sbias[0 * 256 + ncol];
    float bf = sbias[1 * 256 + ncol];
    float bg = sbias[2 * 256 + ncol];
    float bo = sbias[3 * 256 + ncol];
    int m0 = mt * 16 + ((lane >> 4) << 3);            // C/D row base for this lane
#pragma unroll
    for (int r = 0; r < 8; ++r) {
      float ig = sigmoid_f(a0[r] + bi);
      float fg = sigmoid_f(a1[r] + bf);
      float gg = tanh_f   (a2[r] + bg);
      float og = sigmoid_f(a3[r] + bo);
      float c  = fg * cst[task][r] + ig * gg;
      cst[task][r] = c;
      float h  = og * tanh_f(c);
      half_t hh = (half_t)h;
      hout[(m0 + r) * Hsz + ncol] = hh;
      if (gdump) gdump[(size_t)(m0 + r) * Hsz + ncol] = hh;
    }
  }
}

__global__ __launch_bounds__(1024, 1) void rnn_kernel(
    const half_t* __restrict__ xs_enc,   // [128][64][128] f16
    const half_t* __restrict__ xs_dec,   // [127][64][128] f16
    const half_t* eWih0, const half_t* eWhh0, const half_t* eWih1, const half_t* eWhh1,
    const float* ebih0, const float* ebhh0, const float* ebih1, const float* ebhh1,
    const half_t* dWih0, const half_t* dWhh0, const half_t* dWih1, const half_t* dWhh1,
    const float* dbih0, const float* dbhh0, const float* dbih1, const float* dbhh1,
    half_t* __restrict__ h1out)          // [127][64][256] f16
{
  __shared__ half_t sx[Bsz * Esz];        // 16 KB staged x_t
  __shared__ half_t sh0[2][Bsz * Hsz];    // 64 KB double-buffered h (layer 0)
  __shared__ half_t sh1[2][Bsz * Hsz];    // 64 KB double-buffered h (layer 1)
  __shared__ float  sb0[1024];            // 4 KB bias sums, layer 0 (per phase)
  __shared__ float  sb1[1024];            // 4 KB bias sums, layer 1 (per phase)
  int tid  = threadIdx.x;
  int wave = tid >> 5, lane = tid & 31;

  float c0[2][8], c1[2][8];
#pragma unroll
  for (int a = 0; a < 2; ++a)
#pragma unroll
    for (int r = 0; r < 8; ++r) { c0[a][r] = 0.f; c1[a][r] = 0.f; }

  for (int i = tid; i < Bsz * Hsz; i += 1024) {
    sh0[0][i] = (half_t)0.f; sh0[1][i] = (half_t)0.f;
    sh1[0][i] = (half_t)0.f; sh1[1][i] = (half_t)0.f;
  }

  int p = 0;
  for (int ph = 0; ph < 2; ++ph) {
    const half_t* Wih0 = ph ? dWih0 : eWih0;
    const half_t* Whh0 = ph ? dWhh0 : eWhh0;
    const half_t* Wih1 = ph ? dWih1 : eWih1;
    const half_t* Whh1 = ph ? dWhh1 : eWhh1;
    const float*  bi0  = ph ? dbih0 : ebih0;
    const float*  bh0  = ph ? dbhh0 : ebhh0;
    const float*  bi1  = ph ? dbih1 : ebih1;
    const float*  bh1  = ph ? dbhh1 : ebhh1;
    const half_t* xs   = ph ? xs_dec : xs_enc;
    int steps = ph ? (Tsz - 1) : Ssz;

    // time-invariant bias sums -> LDS (tid covers 0..1023 exactly)
    sb0[tid] = bi0[tid] + bh0[tid];
    sb1[tid] = bi1[tid] + bh1[tid];
    __syncthreads();

    for (int t = 0; t < steps; ++t) {
      ((uint4*)sx)[tid] = ((const uint4*)(xs + (size_t)t * (Bsz * Esz)))[tid]; // 16 KB
      if (t + 1 < steps)   // pull next step's x toward the caches while we compute
        __builtin_prefetch(xs + (size_t)(t + 1) * (Bsz * Esz) + tid * 8, 0, 1);
      __syncthreads();
      layer_step<Esz>(sx, sh0[p], Wih0, Whh0, sb0, c0, sh0[p ^ 1],
                      (half_t*)nullptr, wave, lane);
      __syncthreads();
      layer_step<Hsz>(sh0[p ^ 1], sh1[p], Wih1, Whh1, sb1, c1, sh1[p ^ 1],
                      ph ? (h1out + (size_t)t * (Bsz * Hsz)) : (half_t*)nullptr, wave, lane);
      __syncthreads();
      p ^= 1;
    }
  }
}

// ---------------- FC GEMM: [8128,256] x [256,32000] + bias -> out[:,1:,:] ----
// Block tile 64M x 128N; accumulate with WMMA, stage through padded LDS, then
// emit fully coalesced float4 row stores (this stream is the HBM-bound one).
__global__ __launch_bounds__(256, 1) void fc_kernel(
    const half_t* __restrict__ Hx,   // [127*64][256] f16, row = t*64 + b
    const half_t* __restrict__ Wf,   // [32000][256] f16
    const float*  __restrict__ bias, // [32000] f32
    float* __restrict__ out)         // [64][128][32000] f32
{
  __shared__ float so[64][132];      // 33 KB, +4 pad kills bank conflicts
  int tid  = threadIdx.x;
  int wave = tid >> 5, lane = tid & 31;
  int bm = blockIdx.x;               // 0..126  (64 rows each)
  int bn = blockIdx.y;               // 0..249  (128 cols each)
  int mt = wave & 3;
  int nb = (wave >> 2) * 64;         // which 64-col half of the block tile
  int m0 = bm * 64 + mt * 16;

  v8f acc[4] = {};
  for (int k0 = 0; k0 < Hsz; k0 += 32) {
    v16h a = load_frag_a(Hx, Hsz, m0, k0, lane);
#pragma unroll
    for (int j = 0; j < 4; ++j) {
      v16h b = load_frag_b(Wf, Hsz, bn * 128 + nb + j * 16, k0, lane);
      acc[j] = __builtin_amdgcn_wmma_f32_16x16x32_f16(false, a, false, b, (short)0, acc[j], false, false);
    }
  }
  int mloc0 = mt * 16 + ((lane >> 4) << 3);
#pragma unroll
  for (int j = 0; j < 4; ++j) {
    int nloc = nb + j * 16 + (lane & 15);
    float bv = bias[bn * 128 + nloc];
#pragma unroll
    for (int r = 0; r < 8; ++r) so[mloc0 + r][nloc] = acc[j][r] + bv;
  }
  __syncthreads();
  // 64 rows x 128 cols = 2048 float4 groups; 256 threads x 8 iterations
#pragma unroll
  for (int it = 0; it < 8; ++it) {
    int i   = tid + it * 256;
    int row = i >> 5;                // 32 float4-groups per row
    int cg  = (i & 31) << 2;
    int m = bm * 64 + row;
    int tt = m >> 6;                 // decoder step
    int bb = m & 63;                 // batch
    float4 v = *(const float4*)&so[row][cg];
    *(float4*)&out[(size_t)bb * ((size_t)Tsz * Vout) + (size_t)(tt + 1) * Vout
                   + (size_t)bn * 128 + cg] = v;
  }
}

// ---------------- small utility kernels ----------------
__global__ void cvt_f32_to_f16(const float* __restrict__ s, half_t* __restrict__ d, int n) {
  int i = blockIdx.x * 256 + threadIdx.x;
  if (i < n) d[i] = (half_t)s[i];
}

__global__ void gather_enc_kernel(const int* __restrict__ src, const float* __restrict__ emb,
                                  half_t* __restrict__ outx) {
  int i = blockIdx.x * 256 + threadIdx.x;      // < 128*64*128
  int e = i & 127, b = (i >> 7) & 63, t = i >> 13;
  int tok = src[b * Ssz + t];
  outx[i] = (half_t)emb[(size_t)tok * Esz + e];
}

__global__ void gather_dec_kernel(const int* __restrict__ trg, const float* __restrict__ emb,
                                  half_t* __restrict__ outx) {
  int i = blockIdx.x * 256 + threadIdx.x;      // < 127*64*128
  if (i >= 127 * Bsz * Esz) return;
  int e = i & 127, b = (i >> 7) & 63, t = i >> 13;
  int tok = (t == 0) ? 1 : trg[b * Tsz + t];   // SOS then teacher-forced trg[:,t]
  outx[i] = (half_t)emb[(size_t)tok * Esz + e];
}

__global__ void zero_t0_kernel(float* __restrict__ out) {
  int i = blockIdx.x * 256 + threadIdx.x;      // < 64*32000
  if (i < Bsz * Vout) {
    int b = i / Vout, v = i % Vout;
    out[(size_t)b * ((size_t)Tsz * Vout) + v] = 0.f;
  }
}

// ---------------- host launcher ----------------
extern "C" void kernel_launch(void* const* d_in, const int* in_sizes, int n_in,
                              void* d_out, int out_size, void* d_ws, size_t ws_size,
                              hipStream_t stream) {
  (void)in_sizes; (void)n_in; (void)out_size; (void)ws_size;
  const int*   src     = (const int*)  d_in[0];
  const int*   trg     = (const int*)  d_in[1];
  const float* enc_emb = (const float*)d_in[2];
  const float* dec_emb = (const float*)d_in[3];
  const float* eWih0f  = (const float*)d_in[4];
  const float* eWhh0f  = (const float*)d_in[5];
  const float* ebih0   = (const float*)d_in[6];
  const float* ebhh0   = (const float*)d_in[7];
  const float* eWih1f  = (const float*)d_in[8];
  const float* eWhh1f  = (const float*)d_in[9];
  const float* ebih1   = (const float*)d_in[10];
  const float* ebhh1   = (const float*)d_in[11];
  const float* dWih0f  = (const float*)d_in[12];
  const float* dWhh0f  = (const float*)d_in[13];
  const float* dbih0   = (const float*)d_in[14];
  const float* dbhh0   = (const float*)d_in[15];
  const float* dWih1f  = (const float*)d_in[16];
  const float* dWhh1f  = (const float*)d_in[17];
  const float* dbih1   = (const float*)d_in[18];
  const float* dbhh1   = (const float*)d_in[19];
  const float* fcWf    = (const float*)d_in[20];
  const float* fcb     = (const float*)d_in[21];
  float* out = (float*)d_out;

  char* wp = (char*)d_ws;
  auto alloc_h = [&](size_t nhalfs) {
    half_t* p = (half_t*)wp;
    wp += ((nhalfs * sizeof(half_t) + 255) & ~(size_t)255);
    return p;
  };
  half_t* xs_enc = alloc_h((size_t)Ssz * Bsz * Esz);
  half_t* xs_dec = alloc_h((size_t)(Tsz - 1) * Bsz * Esz);
  half_t* eWih0  = alloc_h(1024 * 128);
  half_t* eWhh0  = alloc_h(1024 * 256);
  half_t* eWih1  = alloc_h(1024 * 256);
  half_t* eWhh1  = alloc_h(1024 * 256);
  half_t* dWih0  = alloc_h(1024 * 128);
  half_t* dWhh0  = alloc_h(1024 * 256);
  half_t* dWih1  = alloc_h(1024 * 256);
  half_t* dWhh1  = alloc_h(1024 * 256);
  half_t* fcW    = alloc_h((size_t)Vout * Hsz);
  half_t* h1d    = alloc_h((size_t)(Tsz - 1) * Bsz * Hsz);

  auto cvt = [&](const float* s, half_t* d, int n) {
    cvt_f32_to_f16<<<(n + 255) / 256, 256, 0, stream>>>(s, d, n);
  };
  cvt(eWih0f, eWih0, 1024 * 128);
  cvt(eWhh0f, eWhh0, 1024 * 256);
  cvt(eWih1f, eWih1, 1024 * 256);
  cvt(eWhh1f, eWhh1, 1024 * 256);
  cvt(dWih0f, dWih0, 1024 * 128);
  cvt(dWhh0f, dWhh0, 1024 * 256);
  cvt(dWih1f, dWih1, 1024 * 256);
  cvt(dWhh1f, dWhh1, 1024 * 256);
  cvt(fcWf,   fcW,   Vout * Hsz);

  gather_enc_kernel<<<(Ssz * Bsz * Esz) / 256, 256, 0, stream>>>(src, enc_emb, xs_enc);
  gather_dec_kernel<<<((Tsz - 1) * Bsz * Esz + 255) / 256, 256, 0, stream>>>(trg, dec_emb, xs_dec);
  zero_t0_kernel<<<(Bsz * Vout + 255) / 256, 256, 0, stream>>>(out);

  rnn_kernel<<<1, 1024, 0, stream>>>(xs_enc, xs_dec,
      eWih0, eWhh0, eWih1, eWhh1, ebih0, ebhh0, ebih1, ebhh1,
      dWih0, dWhh0, dWih1, dWhh1, dbih0, dbhh0, dbih1, dbhh1, h1d);

  fc_kernel<<<dim3(Tsz - 1, Vout / 128), 256, 0, stream>>>(h1d, fcW, fcb, out);
}
